// MaskedSelfAttend_9474697855745
// MI455X (gfx1250) — matrix operand order repaired and checked by reading, not compiled, not run on hardware
//
#include <hip/hip_runtime.h>

#define B_ 16
#define L_ 2048
#define H_ 128

typedef __attribute__((ext_vector_type(16))) _Float16 v16h;
typedef __attribute__((ext_vector_type(8)))  _Float16 h8;
typedef __attribute__((ext_vector_type(4)))  _Float16 h4;
typedef __attribute__((ext_vector_type(2)))  _Float16 h2;
typedef __attribute__((ext_vector_type(8)))  float    v8f;
typedef __attribute__((ext_vector_type(4)))  unsigned int u32x4;
typedef __attribute__((ext_vector_type(8)))  int      i32x8;
typedef __attribute__((ext_vector_type(4)))  int      i32x4;

#define SCALE_QK 0.29730177875068026f   // 128^(-1/4): folded into both operands
#define LOG2E    1.4426950408889634f
#define NBIAS2   (-11.541560327111707f) // -8 * log2(e): fixed softmax bias

#if __has_builtin(__builtin_amdgcn_exp2f)
#define EXP2F __builtin_amdgcn_exp2f
#else
#define EXP2F exp2f
#endif

__device__ inline v16h combine16(h8 lo, h8 hih) {
  v16h r;
#pragma unroll
  for (int i = 0; i < 8; ++i) { r[i] = lo[i]; r[8 + i] = hih[i]; }
  return r;
}

// Generic pointers into LDS carry the wave-relative LDS offset in the low 32
// bits (shared aperture lives in the high half).
__device__ inline uint32_t lds_off(const void* p) { return (uint32_t)(size_t)p; }

#if __has_builtin(__builtin_amdgcn_tensor_load_to_lds)
#define USE_TDM 1
// One TDM descriptor per 8KB tile (ISA cdna5 §8.3/8.4 D# layout), 2D mode:
//  g0: [1:0]=count=1 | lds_addr[63:32] | global_addr[120:64] | type[127:126]=2
//  g1: data_size[17:16]=3 (8B units) | tensor_dim0/1 | tile_dim0/1 | stride0
// This toolchain exposes the 6-arg builtin: (g0, g1, g2, g3, g4, cpol);
// the trailing i32x8 is unused for <=2D tiles and passed as zeros.
__device__ inline void tdm_load_2d(uint32_t lds, const void* gaddr,
                                   uint32_t td0, uint32_t td1,
                                   uint32_t tile0, uint32_t tile1,
                                   uint32_t stride0) {
  uint64_t ga = (uint64_t)(size_t)gaddr;
  u32x4 g0;
  g0[0] = 1u;                                   // count=1, user descriptor
  g0[1] = lds;                                  // lds_addr (bytes)
  g0[2] = (uint32_t)ga;                         // global_addr[31:0]
  g0[3] = (uint32_t)(ga >> 32) | 0x80000000u;   // global_addr[56:32] | type=2
  i32x8 g1;
  g1[0] = (int)0x00030000u;                     // data_size=3 (8B), no flags
  g1[1] = (int)((td0 & 0xffffu) << 16);         // tensor_dim0[15:0]
  g1[2] = (int)((td0 >> 16) | ((td1 & 0xffffu) << 16));
  g1[3] = (int)((td1 >> 16) | (tile0 << 16));   // tile_dim0
  g1[4] = (int)(tile1 & 0xffffu);               // tile_dim1 (tile_dim2=0)
  g1[5] = (int)stride0;                         // tensor_dim0_stride[31:0]
  g1[6] = 0;                                    // stride0 hi / stride1 lo
  g1[7] = 0;
  i32x4 gz4 = {0, 0, 0, 0};                     // groups 2/3 unused (2D)
  i32x8 gz8 = {0, 0, 0, 0, 0, 0, 0, 0};
  __builtin_amdgcn_tensor_load_to_lds(g0, g1, gz4, gz4, gz8, 0);
}
#else
#define USE_TDM 0
__device__ inline void async_copy_b128(uint32_t lds, const void* g) {
  asm volatile("global_load_async_to_lds_b128 %0, %1, off"
               :: "v"(lds), "v"(g) : "memory");
}
__device__ inline void wait_async0() {
  asm volatile("s_wait_asynccnt 0x0" ::: "memory");
}
#endif

// ---------------------------------------------------------------------------
// Prep: q (f32) -> qh  = f16 [b][l][h] scaled by 128^(-1/4)
//               -> qht = f16 [b][h][l'] scaled, keys interleaved inside each
//                  32-block: l = 32m + 16t + n  ->  l' = 32m + 2n + t
// ---------------------------------------------------------------------------
__global__ __launch_bounds__(256) void prep_f16_kernel(
    const float* __restrict__ q, _Float16* __restrict__ qh,
    _Float16* __restrict__ qht) {
  __shared__ __align__(16) _Float16 T[128 * 128];  // [l_local][h]
  const int tid = threadIdx.x;
  const int b   = blockIdx.x >> 4;
  const int lt  = blockIdx.x & 15;
  const int l0  = lt * 128;

  const float* src = q  + ((size_t)b * L_ + l0) * H_;
  _Float16*    dst = qh + ((size_t)b * L_ + l0) * H_;
#pragma unroll
  for (int i = 0; i < 16; ++i) {
    int j  = tid + i * 256;            // 4096 float4s = 128x128 tile
    int ll = j >> 5;
    int hh = (j & 31) * 4;
    float4 v = *(const float4*)(src + (size_t)ll * H_ + hh);
    h4 pk;
    pk[0] = (_Float16)(v.x * SCALE_QK);
    pk[1] = (_Float16)(v.y * SCALE_QK);
    pk[2] = (_Float16)(v.z * SCALE_QK);
    pk[3] = (_Float16)(v.w * SCALE_QK);
    *(h4*)(dst + (size_t)ll * H_ + hh) = pk;
    *(h4*)&T[ll * 128 + hh] = pk;
  }
  __syncthreads();
#pragma unroll
  for (int i = 0; i < 16; ++i) {
    int j  = tid + i * 256;            // 4096 h4-chunks of transposed tile
    int h  = j >> 5;
    int lp = (j & 31) * 4;             // base l' of this chunk
    int m  = lp >> 5;
    int lb = lp & 31;
    h4 pk;
#pragma unroll
    for (int e = 0; e < 4; ++e) {
      int le = lb + e;
      int n = le >> 1, t = le & 1;
      int l = (m << 5) | (t << 4) | n; // inverse permutation
      pk[e] = T[l * 128 + h];
    }
    *(h4*)(qht + ((size_t)b * H_ + h) * L_ + l0 + lp) = pk;
  }
}

// ---------------------------------------------------------------------------
// Flash attention: 8 waves/block, one 16-row query tile per wave, 32-key
// blocks streamed through double-buffered LDS via the Tensor Data Mover.
// ---------------------------------------------------------------------------
__global__ __launch_bounds__(256) void MaskedSelfAttend_kernel(
    const _Float16* __restrict__ qh, const _Float16* __restrict__ qht,
    float* __restrict__ out) {
  __shared__ __align__(16) _Float16 Krm[2][32 * H_];   // [key][h]   (S phase)
  __shared__ __align__(16) _Float16 Kt [2][H_ * 32];   // [h][key']  (V phase)
  __shared__ __align__(16) _Float16 Pbuf[8][16 * 32];  // per-wave P [M][key']

  const int tid  = threadIdx.x;
  const int w    = tid >> 5;
  const int lane = tid & 31;
  const int hig  = lane >> 4;
  const int ln   = lane & 15;

  const int b  = blockIdx.x >> 4;
  const int tg = blockIdx.x & 15;
  const int qb = tg * 128 + w * 16;

  const _Float16* qhb  = qh  + (size_t)b * L_ * H_;
  const _Float16* qhtb = qht + (size_t)b * H_ * L_;

  // Q tile as four 16x32 A-fragments, straight from the f16 copy.
  v16h Qf[4];
  {
    const _Float16* rowp = qhb + (size_t)(qb + ln) * H_;
#pragma unroll
    for (int c = 0; c < 4; ++c)
      Qf[c] = combine16(*(const h8*)(rowp + c * 32 + hig * 16),
                        *(const h8*)(rowp + c * 32 + hig * 16 + 8));
  }

  v8f zero = {};
  v8f O[8];
#pragma unroll
  for (int hc = 0; hc < 8; ++hc) O[hc] = zero;
  v8f OL = zero;                       // row-sums of P via WMMA with ones
  v16h ONES;
#pragma unroll
  for (int i = 0; i < 16; ++i) ONES[i] = (_Float16)1.0f;

  auto issue_block = [&](int buf, int kb) {
#if USE_TDM
    // Krm: one contiguous 8KB line; Kt: 128 rows x 64B, row stride 4KB.
    if (w == 0)
      tdm_load_2d(lds_off(&Krm[buf][0]), qhb + (size_t)kb * H_,
                  /*td0=*/1024, /*td1=*/1, /*tile0=*/1024, /*tile1=*/1,
                  /*stride0=*/1024);
    if (w == 1)
      tdm_load_2d(lds_off(&Kt[buf][0]), qhtb + kb,
                  /*td0=*/512, /*td1=*/128, /*tile0=*/8, /*tile1=*/128,
                  /*stride0=*/512);
#else
    uint32_t krm = lds_off(&Krm[buf][0]);
    uint32_t kt  = lds_off(&Kt[buf][0]);
    const char* g0 = (const char*)(qhb + (size_t)kb * H_);
#pragma unroll
    for (int i = 0; i < 2; ++i) {
      int j = tid + i * 256;
      async_copy_b128(krm + j * 16, g0 + j * 16);
    }
#pragma unroll
    for (int i = 0; i < 2; ++i) {
      int j = tid + i * 256;
      int h = j >> 2, c4 = j & 3;
      async_copy_b128(kt + j * 16,
                      (const char*)(qhtb + (size_t)h * L_ + kb) + c4 * 16);
    }
#endif
  };

  const int diagIt = qb >> 5;          // block containing this tile's diagonal
  const int key0   = ln;               // per-lane key column (tile 0 / tile 1)
  issue_block(0, 0);

  for (int it = 0; it < L_ / 32; ++it) {
    const int kb  = it * 32;
    const int cur = it & 1;
#if USE_TDM
    if (w < 2) __builtin_amdgcn_s_wait_tensorcnt(0);
#else
    wait_async0();
#endif
    __syncthreads();                   // tiles visible to all waves
    if (it + 1 < L_ / 32) issue_block(cur ^ 1, kb + 32);

    // ---- S = Q K^T for two 16-key tiles ----
    v8f S0 = zero, S1 = zero;
#pragma unroll
    for (int c = 0; c < 4; ++c) {
      const _Float16* b0p = &Krm[cur][(ln)      * H_ + c * 32 + hig * 16];
      const _Float16* b1p = &Krm[cur][(16 + ln) * H_ + c * 32 + hig * 16];
      v16h Bf0 = combine16(*(const h8*)b0p, *(const h8*)(b0p + 8));
      v16h Bf1 = combine16(*(const h8*)b1p, *(const h8*)(b1p + 8));
      S0 = __builtin_amdgcn_wmma_f32_16x16x32_f16(false, Qf[c], false, Bf0,
                                                  (short)0, S0, false, false);
      S1 = __builtin_amdgcn_wmma_f32_16x16x32_f16(false, Qf[c], false, Bf1,
                                                  (short)0, S1, false, false);
    }

    // ---- max-free softmax numerator; P stored key-interleaved (key'=2n+t) ----
    if (it != diagIt) {                // hot path: no mask compares
#pragma unroll
      for (int r = 0; r < 8; ++r) {
        h2 pp;
        pp[0] = (_Float16)EXP2F(__builtin_fmaf(S0[r], LOG2E, NBIAS2));
        pp[1] = (_Float16)EXP2F(__builtin_fmaf(S1[r], LOG2E, NBIAS2));
        *(h2*)&Pbuf[w][(r + 8 * hig) * 32 + 2 * ln] = pp;
      }
    } else {                           // one block per wave: mask self column
#pragma unroll
      for (int r = 0; r < 8; ++r) {
        int qrow = (qb & 31) + r + 8 * hig;   // row index within 32-key block
        float p0 = EXP2F(__builtin_fmaf(S0[r], LOG2E, NBIAS2));
        float p1 = EXP2F(__builtin_fmaf(S1[r], LOG2E, NBIAS2));
        if (qrow == key0)      p0 = 0.0f;
        if (qrow == key0 + 16) p1 = 0.0f;
        h2 pp; pp[0] = (_Float16)p0; pp[1] = (_Float16)p1;
        *(h2*)&Pbuf[w][(r + 8 * hig) * 32 + 2 * ln] = pp;
      }
    }

    // ---- O += P V ; OL += P 1 (row sums land in matching D-layout slots) ----
    const _Float16* prow = &Pbuf[w][ln * 32];
    v16h Pf = combine16(*(const h8*)(prow + hig * 16),
                        *(const h8*)(prow + hig * 16 + 8));
    OL = __builtin_amdgcn_wmma_f32_16x16x32_f16(false, Pf, false, ONES,
                                                (short)0, OL, false, false);
#pragma unroll
    for (int hc = 0; hc < 8; ++hc) {
      const _Float16* vp = &Kt[cur][(hc * 16 + ln) * 32 + hig * 16];
      v16h Vf = combine16(*(const h8*)vp, *(const h8*)(vp + 8));
      O[hc] = __builtin_amdgcn_wmma_f32_16x16x32_f16(false, Pf, false, Vf,
                                                     (short)0, O[hc], false, false);
    }
    __syncthreads();                   // all reads of buffer `cur` done
  }

  // ---- normalize: out = (O / SCALE_QK) / rowsum ----
#pragma unroll
  for (int r = 0; r < 8; ++r) {
    float inv = 1.0f / (OL[r] * SCALE_QK);
    float* orow = out + ((size_t)b * L_ + qb + r + 8 * hig) * (size_t)H_;
#pragma unroll
    for (int hc = 0; hc < 8; ++hc) orow[hc * 16 + ln] = O[hc][r] * inv;
  }
}

extern "C" void kernel_launch(void* const* d_in, const int* in_sizes, int n_in,
                              void* d_out, int out_size, void* d_ws, size_t ws_size,
                              hipStream_t stream) {
  (void)in_sizes; (void)n_in; (void)out_size; (void)ws_size;
  const float* q = (const float*)d_in[0];
  float* out = (float*)d_out;

  // ws layout: qh f16 [B][L][H] | qht f16 [B][H][L]  (needs 16.8 MB)
  _Float16* qh  = (_Float16*)d_ws;
  _Float16* qht = qh + (size_t)B_ * L_ * H_;

  prep_f16_kernel<<<dim3(B_ * (L_ / 128)), dim3(256), 0, stream>>>(q, qh, qht);
  MaskedSelfAttend_kernel<<<dim3(B_ * (L_ / 128)), dim3(256), 0, stream>>>(
      qh, qht, out);
}
